// LatentSDE_84421877170854
// MI455X (gfx1250) — matrix operand order repaired and checked by reading, not compiled, not run on hardware
//
#include <hip/hip_runtime.h>
#include <hip/hip_bf16.h>
#include <cstdint>

// ---------------------------------------------------------------------------
// CDNA5 (gfx1250) latent-SDE forward.
//   - wave32, WMMA bf16 16x16x32 with f32 accumulation
//   - batch-parallel workgroups, no cross-WG synchronization
//   - weights converted once to bf16 in workspace (L2-resident, 192MB L2)
//   - an opaque zero OFFSET (not pointer) is re-materialized per sequential
//     step: blocks hoisting of t-invariant weight fragments (round-1 spilled
//     them to scratch) while keeping address-space inference intact so weight
//     fetches lower to global_load_b128 instead of flat_load_b128 (round-2).
//   - __builtin_prefetch on next step's streamed rows (global_prefetch_b8).
// ---------------------------------------------------------------------------

typedef __bf16 bf16_t;
typedef bf16_t v16bf __attribute__((ext_vector_type(16)));
typedef float  v8f   __attribute__((ext_vector_type(8)));

union FragU { v16bf v; uint4 q[2]; };

// Model dims
#define BB   64
#define TT   1000
#define NN   256     // neurons
#define HH   256     // GRU hidden
#define LL   64      // latent
#define CC   8       // control
#define KX   288     // D_in=257 padded to mult of 32
#define KP   352     // post in = 328 padded
#define KQ   96      // prior in = 72 padded

// ---------------- WMMA fragment helpers (layouts per CDNA5 ISA) --
__device__ __forceinline__ v16bf ld_a(const bf16_t* base, int ld, int k0, int lane) {
  int r = lane & 15, h = (lane >> 4) & 1;
  const bf16_t* p = base + (size_t)r * ld + k0 + 8 * h;
  FragU f;
  f.q[0] = *(const uint4*)(p);
  f.q[1] = *(const uint4*)(p + 16);
  return f.v;
}
__device__ __forceinline__ v16bf ld_b(const bf16_t* base, int ld, int k0, int lane) {
  int c = lane & 15, h = (lane >> 4) & 1;
  const bf16_t* p = base + (size_t)c * ld + k0 + 8 * h;
  FragU f;
  f.q[0] = *(const uint4*)(p);
  f.q[1] = *(const uint4*)(p + 16);
  return f.v;
}
__device__ __forceinline__ v8f wmma_bf(v16bf a, v16bf b, v8f c) {
  return __builtin_amdgcn_wmma_f32_16x16x32_bf16(false, a, false, b, (short)0, c,
                                                 false, false);
}
__device__ __forceinline__ v8f vzero() {
  v8f c;
#pragma unroll
  for (int i = 0; i < 8; ++i) c[i] = 0.0f;
  return c;
}
// Opaque zero offset, re-materialized wherever called: loads whose address
// includes it cannot be hoisted out of the enclosing loop, but the pointer
// chain stays "kernarg + index" so loads still lower to global_load (saddr).
__device__ __forceinline__ size_t opq0() {
  size_t z = 0;
  asm volatile("" : "+s"(z));
  return z;
}

__device__ __forceinline__ float sigmoidf_(float x) { return 1.0f / (1.0f + __expf(-x)); }
__device__ __forceinline__ float softplusf_(float x) {
  return fmaxf(x, 0.0f) + log1pf(__expf(-fabsf(x)));
}

// ---------------- weight pack: f32 -> bf16 with zero padding ----------------
__global__ void k_pack(const float* __restrict__ src, bf16_t* __restrict__ dst,
                       int sR, int sK, int dR, int dK) {
  int i = blockIdx.x * 256 + threadIdx.x;
  if (i >= dR * dK) return;
  int r = i / dK, k = i - r * dK;
  float v = (r < sR && k < sK) ? src[(size_t)r * sK + k] : 0.0f;
  dst[i] = (bf16_t)v;
}

// ---------------- GRU encoder (reverse-time), 16 batch cols per WG ----------
__global__ __launch_bounds__(256) void k_encoder(
    const bf16_t* __restrict__ Wih, const bf16_t* __restrict__ Whh,
    const float* __restrict__ bih, const float* __restrict__ bhh,
    const float* __restrict__ ts, const float* __restrict__ ys,
    bf16_t* __restrict__ ctx) {
  __shared__ alignas(16) bf16_t xlds[16 * KX];
  __shared__ alignas(16) bf16_t hbf[16 * HH];
  __shared__ alignas(16) float  hf[16 * HH];
  __shared__ float bihL[768];
  __shared__ float bhhL[768];
  const int tid = threadIdx.x, w = tid >> 5, lane = tid & 31;
  const int b0 = blockIdx.x * 16;
  const int col = lane & 15, half = (lane >> 4) & 1;

  for (int i = tid; i < 16 * KX; i += 256) xlds[i] = (bf16_t)0.0f;
  for (int i = tid; i < 16 * HH; i += 256) { hbf[i] = (bf16_t)0.0f; hf[i] = 0.0f; }
  for (int i = tid; i < 768; i += 256) { bihL[i] = bih[i]; bhhL[i] = bhh[i]; }
  __syncthreads();

#pragma unroll 1
  for (int t = TT - 1; t >= 0; --t) {
    const bf16_t* WihT = Wih + opq0();
    const bf16_t* WhhT = Whh + opq0();
    // prefetch next step's ys rows (16 cols x 1KB) toward the caches
    if (t > 0 && tid < 128) {
      int c = tid >> 3, seg = tid & 7;
      __builtin_prefetch((const char*)&ys[(((size_t)(b0 + c)) * TT + (t - 1)) * NN] +
                             seg * 128, 0, 3);
    }
    // stage x_t = [ts[t], ys[b,t,:]] as bf16, per-column contiguous
    float tv = ts[t];
#pragma unroll 1
    for (int i = tid; i < 16 * 257; i += 256) {
      int c = i / 257, k = i - c * 257;
      float v = (k == 0) ? tv : ys[(((size_t)(b0 + c)) * TT + t) * NN + (k - 1)];
      xlds[c * KX + k] = (bf16_t)v;
    }
    __syncthreads();

    // wave w owns gate tiles {mr, 16+mr, 32+mr} for mr = 2w, 2w+1.
    // r/z gates take Wih*x + Whh*h summed -> shared accumulator.
    v8f ar[2], az[2], ain[2], ahn[2];
#pragma unroll
    for (int j = 0; j < 2; ++j) { ar[j] = vzero(); az[j] = vzero(); ain[j] = vzero(); ahn[j] = vzero(); }

#pragma unroll 1
    for (int k0 = 0; k0 < KX; k0 += 32) {
      v16bf bx = ld_b(xlds, KX, k0, lane);
#pragma unroll
      for (int j = 0; j < 2; ++j) {
        int mr = 2 * w + j;
        v16bf a0 = ld_a(WihT + (size_t)(mr) * 16 * KX, KX, k0, lane);
        ar[j] = wmma_bf(a0, bx, ar[j]);
        v16bf a1 = ld_a(WihT + (size_t)(16 + mr) * 16 * KX, KX, k0, lane);
        az[j] = wmma_bf(a1, bx, az[j]);
        v16bf a2 = ld_a(WihT + (size_t)(32 + mr) * 16 * KX, KX, k0, lane);
        ain[j] = wmma_bf(a2, bx, ain[j]);
      }
    }
#pragma unroll 1
    for (int k0 = 0; k0 < HH; k0 += 32) {
      v16bf bh = ld_b(hbf, HH, k0, lane);
#pragma unroll
      for (int j = 0; j < 2; ++j) {
        int mr = 2 * w + j;
        v16bf a0 = ld_a(WhhT + (size_t)(mr) * 16 * HH, HH, k0, lane);
        ar[j] = wmma_bf(a0, bh, ar[j]);
        v16bf a1 = ld_a(WhhT + (size_t)(16 + mr) * 16 * HH, HH, k0, lane);
        az[j] = wmma_bf(a1, bh, az[j]);
        v16bf a2 = ld_a(WhhT + (size_t)(32 + mr) * 16 * HH, HH, k0, lane);
        ahn[j] = wmma_bf(a2, bh, ahn[j]);
      }
    }
    // gates + h update (rows owned exclusively by this wave)
    float hnew[2][8];
#pragma unroll
    for (int j = 0; j < 2; ++j) {
      int rb = (2 * w + j) * 16 + 8 * half;
#pragma unroll
      for (int v = 0; v < 8; ++v) {
        int row = rb + v;
        float r = sigmoidf_(ar[j][v] + bihL[row] + bhhL[row]);
        float z = sigmoidf_(az[j][v] + bihL[256 + row] + bhhL[256 + row]);
        float n = tanhf(ain[j][v] + bihL[512 + row] +
                        r * (ahn[j][v] + bhhL[512 + row]));
        float hp = hf[col * HH + row];
        hnew[j][v] = (1.0f - z) * n + z * hp;
      }
    }
    __syncthreads();  // all waves done reading h
#pragma unroll
    for (int j = 0; j < 2; ++j) {
      int rb = (2 * w + j) * 16 + 8 * half;
#pragma unroll
      for (int v = 0; v < 8; ++v) {
        hf[col * HH + rb + v] = hnew[j][v];
        hbf[col * HH + rb + v] = (bf16_t)hnew[j][v];
      }
    }
    __syncthreads();
    // ctx[t] = h  (bf16, per-(b,t) contiguous for later B-fragment loads)
#pragma unroll 1
    for (int i = tid; i < 16 * HH; i += 256) {
      int c = i >> 8, k = i & 255;
      ctx[(((size_t)(b0 + c)) * TT + t) * HH + k] = hbf[i];
    }
  }
}

// ---------------- y0 = W_h2l @ ctx[:,0,:] + b ------------------------------
__global__ void k_y0(const float* __restrict__ Wh, const float* __restrict__ bh,
                     const bf16_t* __restrict__ ctx, float* __restrict__ y0) {
  int b = blockIdx.x, r = threadIdx.x;  // 64 blocks x 64 threads
  const bf16_t* c = ctx + (size_t)b * TT * HH;
  const float* wr = Wh + (size_t)r * HH;
  float s = bh[r];
  for (int k = 0; k < HH; ++k) s += wr[k] * (float)c[k];
  y0[b * LL + r] = s;
}

// ---------------- shared MLP layer helpers (SDE + decode) -------------------
// 256-row layer: wave w computes tiles 2w,2w+1; tanh -> bf16 LDS out[16][256]
// biasL points to an LDS copy (reads cannot be hoisted across barriers).
__device__ __forceinline__ void layer256_tanh(const bf16_t* W, int Kpad,
                                              const float* biasL,
                                              const bf16_t* in, int ld,
                                              int w, int lane, bf16_t* out) {
  v8f acc[2];
#pragma unroll
  for (int j = 0; j < 2; ++j) acc[j] = vzero();
#pragma unroll 1
  for (int k0 = 0; k0 < Kpad; k0 += 32) {
    v16bf b = ld_b(in, ld, k0, lane);
#pragma unroll
    for (int j = 0; j < 2; ++j) {
      v16bf a = ld_a(W + (size_t)(2 * w + j) * 16 * Kpad, Kpad, k0, lane);
      acc[j] = wmma_bf(a, b, acc[j]);
    }
  }
  int col = lane & 15, h = (lane >> 4) & 1;
#pragma unroll
  for (int j = 0; j < 2; ++j) {
    int rb = (2 * w + j) * 16 + 8 * h;
#pragma unroll
    for (int v = 0; v < 8; ++v)
      out[col * 256 + rb + v] = (bf16_t)tanhf(acc[j][v] + biasL[rb + v]);
  }
}
// 64-row output layer (waves 0..3): linear -> f32 LDS out[16][64]
__device__ __forceinline__ void layer64_f32(const bf16_t* W, const float* biasL,
                                            const bf16_t* in, int w, int lane,
                                            float* out) {
  v8f acc = vzero();
#pragma unroll 1
  for (int k0 = 0; k0 < 256; k0 += 32) {
    v16bf b = ld_b(in, 256, k0, lane);
    v16bf a = ld_a(W + (size_t)w * 16 * 256, 256, k0, lane);
    acc = wmma_bf(a, b, acc);
  }
  int col = lane & 15, h = (lane >> 4) & 1;
  int rb = w * 16 + 8 * h;
#pragma unroll
  for (int v = 0; v < 8; ++v) out[col * 64 + rb + v] = acc[v] + biasL[rb + v];
}

// ---------------- SDE Euler-Maruyama solve, 16 batch cols per WG ------------
__global__ __launch_bounds__(256) void k_sde(
    const bf16_t* __restrict__ pW1, const float* __restrict__ pb1,
    const bf16_t* __restrict__ pW2, const float* __restrict__ pb2,
    const bf16_t* __restrict__ pW3, const float* __restrict__ pb3,
    const bf16_t* __restrict__ qW1, const float* __restrict__ qb1,
    const bf16_t* __restrict__ qW2, const float* __restrict__ qb2,
    const bf16_t* __restrict__ qW3, const float* __restrict__ qb3,
    const float* __restrict__ sigma, const float* __restrict__ ts,
    const float* __restrict__ us, const float* __restrict__ dWm,
    const bf16_t* __restrict__ ctx, const float* __restrict__ y0,
    bf16_t* __restrict__ zs, float* __restrict__ logpq) {
  __shared__ alignas(16) bf16_t xp[16 * KP];
  __shared__ alignas(16) bf16_t xq[16 * KQ];
  __shared__ alignas(16) bf16_t act0[16 * 256];
  __shared__ alignas(16) bf16_t act1[16 * 256];
  __shared__ float zf[16 * 64];
  __shared__ float fpost[16 * 64];
  __shared__ float fprior[16 * 64];
  __shared__ float klterm[16 * 64];
  __shared__ float klsh[16];
  __shared__ float biasL[256 + 256 + 64 + 256 + 256 + 64];  // pb1,pb2,pb3,qb1,qb2,qb3
  const int tid = threadIdx.x, w = tid >> 5, lane = tid & 31;
  const int b0 = blockIdx.x * 16;
  const float dt = ts[1] - ts[0];
  const float sqdt = sqrtf(dt);
  float* pb1L = biasL;
  float* pb2L = biasL + 256;
  float* pb3L = biasL + 512;
  float* qb1L = biasL + 576;
  float* qb2L = biasL + 832;
  float* qb3L = biasL + 1088;

  for (int i = tid; i < 16 * KP; i += 256) xp[i] = (bf16_t)0.0f;
  for (int i = tid; i < 16 * KQ; i += 256) xq[i] = (bf16_t)0.0f;
  for (int i = tid; i < 16 * 64; i += 256)
    zf[i] = y0[(size_t)(b0 + (i >> 6)) * LL + (i & 63)];
  if (tid < 16) klsh[tid] = 0.0f;
  { pb1L[tid] = pb1[tid]; pb2L[tid] = pb2[tid]; qb1L[tid] = qb1[tid]; qb2L[tid] = qb2[tid]; }
  if (tid < 64) { pb3L[tid] = pb3[tid]; qb3L[tid] = qb3[tid]; }
  __syncthreads();

#pragma unroll 1
  for (int t = 0; t < TT; ++t) {
    const bf16_t* pW1T = pW1 + opq0();
    const bf16_t* pW2T = pW2 + opq0();
    const bf16_t* pW3T = pW3 + opq0();
    const bf16_t* qW1T = qW1 + opq0();
    const bf16_t* qW2T = qW2 + opq0();
    const bf16_t* qW3T = qW3 + opq0();
    // prefetch next step's ctx rows (16 cols x 512B) + dW rows (256B)
    if (t + 1 < TT && tid < 64) {
      int c = tid >> 2, seg = tid & 3;
      __builtin_prefetch((const char*)&ctx[(((size_t)(b0 + c)) * TT + (t + 1)) * HH] +
                             seg * 128, 0, 3);
      if (seg < 2)
        __builtin_prefetch((const char*)&dWm[(((size_t)(b0 + c)) * TT + (t + 1)) * LL] +
                               seg * 128, 0, 3);
    }
    // zs[t] = z before step t; kls[-1] = kl before step T-1
#pragma unroll 1
    for (int i = tid; i < 16 * 64; i += 256) {
      int c = i >> 6, k = i & 63;
      zs[(((size_t)(b0 + c)) * TT + t) * LL + k] = (bf16_t)zf[i];
    }
    if (t == TT - 1 && tid < 16) logpq[b0 + tid] = klsh[tid];
    // stage post input [z | ctx | u] and prior input [z | u]
#pragma unroll 1
    for (int i = tid; i < 16 * 64; i += 256) {
      int c = i >> 6, k = i & 63;
      bf16_t zb = (bf16_t)zf[i];
      xp[c * KP + k] = zb;
      xq[c * KQ + k] = zb;
    }
#pragma unroll 1
    for (int i = tid; i < 16 * 256; i += 256) {
      int c = i >> 8, k = i & 255;
      xp[c * KP + 64 + k] = ctx[(((size_t)(b0 + c)) * TT + t) * HH + k];
    }
    if (tid < 16 * 8) {
      int c = tid >> 3, k = tid & 7;
      float u = us[(((size_t)(b0 + c)) * TT + t) * CC + k];
      xp[c * KP + 320 + k] = (bf16_t)u;
      xq[c * KQ + 64 + k] = (bf16_t)u;
    }
    __syncthreads();
    // posterior drift MLP
    layer256_tanh(pW1T, KP, pb1L, xp, KP, w, lane, act0);
    __syncthreads();
    layer256_tanh(pW2T, 256, pb2L, act0, 256, w, lane, act1);
    __syncthreads();
    if (w < 4) layer64_f32(pW3T, pb3L, act1, w, lane, fpost);
    __syncthreads();
    // prior drift MLP
    layer256_tanh(qW1T, KQ, qb1L, xq, KQ, w, lane, act0);
    __syncthreads();
    layer256_tanh(qW2T, 256, qb2L, act0, 256, w, lane, act1);
    __syncthreads();
    if (w < 4) layer64_f32(qW3T, qb3L, act1, w, lane, fprior);
    __syncthreads();
    // Euler-Maruyama update + KL terms (deterministic reduction, no atomics)
#pragma unroll 1
    for (int i = tid; i < 16 * 64; i += 256) {
      int c = i >> 6, k = i & 63;
      float g = softplusf_(sigma[k]);
      float fp = fpost[i], fq = fprior[i];
      float u = (fp - fq) / g;
      klterm[i] = 0.5f * u * u * dt;
      float dw = dWm[(((size_t)(b0 + c)) * TT + t) * LL + k];
      zf[i] = zf[i] + fp * dt + g * sqdt * dw;
    }
    __syncthreads();
    if (tid < 16) {
      float s = 0.0f;
      for (int k = 0; k < 64; ++k) s += klterm[tid * 64 + k];
      klsh[tid] += s;
    }
    __syncthreads();
  }
}

// ---------------- decode: hidden -> rates (softplus) + behavior -------------
#define DEC_TPB 8
__global__ __launch_bounds__(256) void k_decode(
    const bf16_t* __restrict__ Wl, const float* __restrict__ bl,
    const bf16_t* __restrict__ Wr, const float* __restrict__ br,
    const bf16_t* __restrict__ Wb, const float* __restrict__ bbv,
    const bf16_t* __restrict__ zs, float* __restrict__ rates,
    float* __restrict__ beh) {
  __shared__ alignas(16) bf16_t act[16 * 256];
  __shared__ float blL[256], brL[256], bbL[4];
  const int tid = threadIdx.x, w = tid >> 5, lane = tid & 31;
  const int col = lane & 15, h = (lane >> 4) & 1;
  { blL[tid] = bl[tid]; brL[tid] = br[tid]; }
  if (tid < 4) bbL[tid] = bbv[tid];
  __syncthreads();
#pragma unroll 1
  for (int rt = blockIdx.x * DEC_TPB; rt < blockIdx.x * DEC_TPB + DEC_TPB; ++rt) {
    const bf16_t* WlT = Wl + opq0();
    const bf16_t* WrT = Wr + opq0();
    const bf16_t* WbT = Wb + opq0();
    size_t rowbase = (size_t)rt * 16;
    {  // hidden = Wl(256x64) @ z  (B fragment straight from global zs)
      v8f acc[2];
#pragma unroll
      for (int j = 0; j < 2; ++j) acc[j] = vzero();
#pragma unroll 1
      for (int k0 = 0; k0 < 64; k0 += 32) {
        v16bf b = ld_b(zs + rowbase * LL, LL, k0, lane);
#pragma unroll
        for (int j = 0; j < 2; ++j) {
          v16bf a = ld_a(WlT + (size_t)(2 * w + j) * 16 * 64, 64, k0, lane);
          acc[j] = wmma_bf(a, b, acc[j]);
        }
      }
#pragma unroll
      for (int j = 0; j < 2; ++j) {
        int rb = (2 * w + j) * 16 + 8 * h;
#pragma unroll
        for (int v = 0; v < 8; ++v)
          act[col * 256 + rb + v] = (bf16_t)(acc[j][v] + blL[rb + v]);
      }
    }
    __syncthreads();
    {  // rates = softplus(Wr @ hidden + br), packed float4 stores
      v8f acc[2];
#pragma unroll
      for (int j = 0; j < 2; ++j) acc[j] = vzero();
#pragma unroll 1
      for (int k0 = 0; k0 < 256; k0 += 32) {
        v16bf b = ld_b(act, 256, k0, lane);
#pragma unroll
        for (int j = 0; j < 2; ++j) {
          v16bf a = ld_a(WrT + (size_t)(2 * w + j) * 16 * 256, 256, k0, lane);
          acc[j] = wmma_bf(a, b, acc[j]);
        }
      }
#pragma unroll
      for (int j = 0; j < 2; ++j) {
        int rb = (2 * w + j) * 16 + 8 * h;
        float* p = rates + (rowbase + col) * NN + rb;
        float4 o0 = make_float4(softplusf_(acc[j][0] + brL[rb + 0]),
                                softplusf_(acc[j][1] + brL[rb + 1]),
                                softplusf_(acc[j][2] + brL[rb + 2]),
                                softplusf_(acc[j][3] + brL[rb + 3]));
        float4 o1 = make_float4(softplusf_(acc[j][4] + brL[rb + 4]),
                                softplusf_(acc[j][5] + brL[rb + 5]),
                                softplusf_(acc[j][6] + brL[rb + 6]),
                                softplusf_(acc[j][7] + brL[rb + 7]));
        *(float4*)(p) = o0;
        *(float4*)(p + 4) = o1;
      }
    }
    if (w == 0) {  // beh = Wb(4->pad16 x 256) @ hidden + bb
      v8f acc = vzero();
#pragma unroll 1
      for (int k0 = 0; k0 < 256; k0 += 32) {
        v16bf b = ld_b(act, 256, k0, lane);
        v16bf a = ld_a(WbT, 256, k0, lane);
        acc = wmma_bf(a, b, acc);
      }
      if (h == 0) {
#pragma unroll
        for (int v = 0; v < 4; ++v)
          beh[(rowbase + col) * 4 + v] = acc[v] + bbL[v];
      }
    }
    __syncthreads();
  }
}

// ---------------------------------------------------------------------------
extern "C" void kernel_launch(void* const* d_in, const int* in_sizes, int n_in,
                              void* d_out, int out_size, void* d_ws, size_t ws_size,
                              hipStream_t stream) {
  (void)in_sizes; (void)n_in; (void)out_size; (void)ws_size;
  // d_in order (insertion-order flatten of setup_inputs):
  //  0 enc.Wih(768x257) 1 enc.Whh(768x256) 2 bih(768) 3 bhh(768)
  //  4 h2l.W(64x256) 5 h2l.b(64) 6 l2h.W(256x64) 7 l2h.b(256)
  //  8 prior.W1(256x72) 9 prior.b1 10 prior.W2(256x256) 11 prior.b2
  // 12 prior.W3(64x256) 13 prior.b3 14 post.W1(256x328) 15 post.b1
  // 16 post.W2(256x256) 17 post.b2 18 post.W3(64x256) 19 post.b3
  // 20 sigma(64) 21 Wr(256x256) 22 br(256) 23 Wb(4x256) 24 bb(4)
  // 25 ts(1000) 26 ys(64x1000x256) 27 us(64x1000x8) 28 dW(64x1000x64)
  char* ws = (char*)d_ws;
  size_t off = 0;
  auto take = [&](size_t bytes) {
    size_t o = off;
    off += (bytes + 255) & ~(size_t)255;
    return o;
  };
  size_t oWih = take((size_t)768 * KX * 2);
  size_t oWhh = take((size_t)768 * HH * 2);
  size_t oPW1 = take((size_t)256 * KP * 2);
  size_t oPW2 = take((size_t)256 * 256 * 2);
  size_t oPW3 = take((size_t)64 * 256 * 2);
  size_t oQW1 = take((size_t)256 * KQ * 2);
  size_t oQW2 = take((size_t)256 * 256 * 2);
  size_t oQW3 = take((size_t)64 * 256 * 2);
  size_t oWl  = take((size_t)256 * 64 * 2);
  size_t oWr  = take((size_t)256 * 256 * 2);
  size_t oWb  = take((size_t)16 * 256 * 2);
  size_t oCtx = take((size_t)BB * TT * HH * 2);
  size_t oZs  = take((size_t)BB * TT * LL * 2);
  size_t oY0  = take((size_t)BB * LL * 4);

  auto pack = [&](int idx, size_t o, int sR, int sK, int dR, int dK) {
    int tot = dR * dK;
    k_pack<<<(tot + 255) / 256, 256, 0, stream>>>(
        (const float*)d_in[idx], (bf16_t*)(ws + o), sR, sK, dR, dK);
  };
  pack(0, oWih, 768, 257, 768, KX);
  pack(1, oWhh, 768, 256, 768, 256);
  pack(14, oPW1, 256, 328, 256, KP);
  pack(16, oPW2, 256, 256, 256, 256);
  pack(18, oPW3, 64, 256, 64, 256);
  pack(8, oQW1, 256, 72, 256, KQ);
  pack(10, oQW2, 256, 256, 256, 256);
  pack(12, oQW3, 64, 256, 64, 256);
  pack(6, oWl, 256, 64, 256, 64);
  pack(21, oWr, 256, 256, 256, 256);
  pack(23, oWb, 4, 256, 16, 256);

  const float* ts = (const float*)d_in[25];
  const float* ys = (const float*)d_in[26];
  const float* us = (const float*)d_in[27];
  const float* dWm = (const float*)d_in[28];

  float* out = (float*)d_out;
  float* out_rates = out;                                  // B*T*256
  float* out_logpq = out + (size_t)BB * TT * NN;           // B
  float* out_beh = out_logpq + BB;                         // B*T*4

  k_encoder<<<BB / 16, 256, 0, stream>>>(
      (const bf16_t*)(ws + oWih), (const bf16_t*)(ws + oWhh),
      (const float*)d_in[2], (const float*)d_in[3], ts, ys,
      (bf16_t*)(ws + oCtx));

  k_y0<<<BB, LL, 0, stream>>>((const float*)d_in[4], (const float*)d_in[5],
                              (const bf16_t*)(ws + oCtx), (float*)(ws + oY0));

  k_sde<<<BB / 16, 256, 0, stream>>>(
      (const bf16_t*)(ws + oPW1), (const float*)d_in[15],
      (const bf16_t*)(ws + oPW2), (const float*)d_in[17],
      (const bf16_t*)(ws + oPW3), (const float*)d_in[19],
      (const bf16_t*)(ws + oQW1), (const float*)d_in[9],
      (const bf16_t*)(ws + oQW2), (const float*)d_in[11],
      (const bf16_t*)(ws + oQW3), (const float*)d_in[13],
      (const float*)d_in[20], ts, us, dWm, (const bf16_t*)(ws + oCtx),
      (const float*)(ws + oY0), (bf16_t*)(ws + oZs), out_logpq);

  int ntiles = (BB * TT) / 16;  // 4000
  k_decode<<<ntiles / DEC_TPB, 256, 0, stream>>>(
      (const bf16_t*)(ws + oWl), (const float*)d_in[7],
      (const bf16_t*)(ws + oWr), (const float*)d_in[22],
      (const bf16_t*)(ws + oWb), (const float*)d_in[24],
      (const bf16_t*)(ws + oZs), out_rates, out_beh);
}
